// TinyGAT_25194278158979
// MI455X (gfx1250) — matrix operand order repaired and checked by reading, not compiled, not run on hardware
//
#include <hip/hip_runtime.h>
#include <math.h>

// ---------------- CDNA5 WMMA / TDM types ----------------
typedef __attribute__((ext_vector_type(16))) __bf16 v16bf;
typedef __attribute__((ext_vector_type(8)))  float  v8f;
typedef __attribute__((ext_vector_type(4)))  unsigned int u32x4;
typedef __attribute__((ext_vector_type(8)))  int i32x8;
typedef __attribute__((ext_vector_type(4)))  int i32x4;

union FragBF { uint4 u[2]; v16bf v; };

#if __has_builtin(__builtin_amdgcn_tensor_load_to_lds) && \
    __has_builtin(__builtin_amdgcn_s_wait_tensorcnt)
#define TDM_OK 1
#else
#define TDM_OK 0
#endif

__device__ __forceinline__ unsigned short f32_to_bf16(float f) {
  unsigned int u = __float_as_uint(f);
  u += 0x7FFFu + ((u >> 16) & 1u);          // round-to-nearest-even
  return (unsigned short)(u >> 16);
}

// float atomic max via signed/unsigned integer atomics (init to -inf).
__device__ __forceinline__ void atomicMaxF32(float* addr, float v) {
  if (v >= 0.0f) atomicMax((int*)addr, __float_as_int(v));
  else           atomicMin((unsigned int*)addr, __float_as_uint(v));
}

// ---------------- elementwise helpers ----------------
__global__ void fill_f32_kernel(float* p, float v, long long n) {
  long long i = (long long)blockIdx.x * blockDim.x + threadIdx.x;
  if (i < n) p[i] = v;
}

__global__ void cvt_bf16_kernel(const float* __restrict__ in,
                                unsigned short* __restrict__ out, long long n) {
  long long i = (long long)blockIdx.x * blockDim.x + threadIdx.x;
  if (i < n) out[i] = f32_to_bf16(in[i]);
}

// Pack W [K,N] row-major f32 -> bf16 fragment layout [K/32][N][32]
// so a B-fragment lane reads two runs of 8 bf16, at k=8h and 16+8h (two B128 loads), and a
// block's (K-step x 256-column) tile is one contiguous 16KB region for TDM.
__global__ void pack_w_kernel(const float* __restrict__ W,
                              unsigned short* __restrict__ Wp, int K, int N) {
  int idx = blockIdx.x * blockDim.x + threadIdx.x;
  if (idx >= K * N) return;
  int kk  = idx & 31;
  int t   = idx >> 5;
  int n   = t % N;
  int kt  = t / N;
  int k   = kt * 32 + kk;
  Wp[idx] = f32_to_bf16(W[(long long)k * N + n]);
}

// ---------------- bf16 WMMA GEMM: C[M,N] = A[M,K] * B[K,N] ----------------
// Block = 256 threads = 8 waves arranged 2(M) x 4(N): block tile 32x256.
// Each wave: 16x64 output = 4 accumulators sharing one A fragment.
// B tile (32K x 256N bf16 = 16KB, contiguous in packed layout) is staged
// into LDS once per K-step by the Tensor Data Mover (wave 0), then each
// lane reads its B fragments with ds_load_b128.
__global__ void __launch_bounds__(256)
gemm_bf16_wmma_kernel(const unsigned short* __restrict__ A,
                      const unsigned short* __restrict__ Bp,
                      float* __restrict__ C, int M, int N, int K) {
  __shared__ unsigned short ldsB[8192];   // 16KB = 256 cols x 32 K-values

  const int tid  = threadIdx.x;
  const int lane = tid & 31;
  const int wave = tid >> 5;
  const int wm   = wave >> 2;             // 0..1: M sub-tile
  const int wq   = wave & 3;              // 0..3: N quarter (64 cols)
  const int r    = lane & 15;
  const int hi   = lane >> 4;

  const int mBase = blockIdx.x * 32;
  const int nBase = blockIdx.y * 256;
  const int aRow  = mBase + wm * 16 + r;

  v8f acc[4] = {{}, {}, {}, {}};

  for (int k = 0; k < K; k += 32) {
    const unsigned short* gB = Bp + ((long long)(k >> 5) * N + nBase) * 32;
#if TDM_OK
    if (wave == 0) {
      // Tensor DMA descriptor (ISA ch.8): 1D copy of 8192 bf16 (16KB).
      unsigned long long ga = (unsigned long long)gB;
      unsigned la = (unsigned)(unsigned long long)&ldsB[0];
      const unsigned td0 = 8192u;         // tensor_dim0 == tile_dim0 (elements)
      u32x4 g0;
      g0[0] = 1u;                          // count=1, is_restore=0, gather=0
      g0[1] = la;                          // lds_addr [63:32]
      g0[2] = (unsigned)ga;                // global_addr [95:64]
      g0[3] = (unsigned)((ga >> 32) & 0x01FFFFFFu) | (2u << 30); // addr hi | type=2
      i32x8 g1;
      g1[0] = 0x00010000;                  // data_size=2B, no mask/pad
      g1[1] = (int)((td0 & 0xFFFFu) << 16);            // tensor_dim0 lo
      g1[2] = (int)((td0 >> 16) & 0xFFFFu) | (1 << 16); // tensor_dim0 hi | dim1=1
      g1[3] = (int)((td0 & 0xFFFFu) << 16);            // tile_dim0
      g1[4] = 1;                           // tile_dim1 = 1
      g1[5] = (int)td0;                    // tensor_dim0_stride
      g1[6] = 0;
      g1[7] = 0;
      i32x4 gz4 = {0, 0, 0, 0};
      i32x8 gz8 = {0, 0, 0, 0, 0, 0, 0, 0};
      // 6-arg form (this toolchain): (g0, g1, g2, g3, g4, cpol)
      __builtin_amdgcn_tensor_load_to_lds(g0, g1, gz4, gz4, gz8, 0);
      __builtin_amdgcn_s_wait_tensorcnt(0);
    }
#else
    for (int t = tid; t < 4096; t += 256)
      ((unsigned int*)ldsB)[t] = ((const unsigned int*)gB)[t];
#endif
    __syncthreads();

    // A fragment (ISA 16-bit A 16x32 layout), direct from global:
    FragBF a;
    const unsigned short* ap = A + (long long)aRow * K + k + hi * 8;
    a.u[0] = *(const uint4*)(ap);
    a.u[1] = *(const uint4*)(ap + 16);

#pragma unroll
    for (int j = 0; j < 4; ++j) {
      FragBF b;
      const unsigned short* bp = &ldsB[(wq * 64 + j * 16 + r) * 32 + hi * 8];
      b.u[0] = *(const uint4*)(bp);
      b.u[1] = *(const uint4*)(bp + 16);
      acc[j] = __builtin_amdgcn_wmma_f32_16x16x32_bf16(
          false, a.v, false, b.v, (short)0, acc[j], false, false);
    }
    __syncthreads();
  }

  // C/D layout: lane column = tile_col + r; VGPR v holds row v + 8*hi.
  float* cbase = C + (long long)(mBase + wm * 16 + hi * 8) * N + nBase + wq * 64 + r;
#pragma unroll
  for (int j = 0; j < 4; ++j) {
    float* cp = cbase + j * 16;
#pragma unroll
    for (int v = 0; v < 8; ++v) cp[(long long)v * N] = acc[j][v];
  }
}

// ---------------- attention coefficients ----------------
__global__ void attn_coeff_kernel(const float* __restrict__ h,
                                  const float* __restrict__ att_src,
                                  const float* __restrict__ att_dst,
                                  float* __restrict__ a_src,
                                  float* __restrict__ a_dst,
                                  int N, int H, int C) {
  int i = blockIdx.x * blockDim.x + threadIdx.x;
  if (i >= N * H) return;
  int n = i / H, hh = i % H;
  const float* hp = h + (long long)n * H * C + (long long)hh * C;
  float ss = 0.f, sd = 0.f;
  for (int c = 0; c < C; ++c) {
    float v = hp[c];
    ss += v * att_src[hh * C + c];
    sd += v * att_dst[hh * C + c];
  }
  a_src[i] = ss;
  a_dst[i] = sd;
}

// ---------------- edge passes (E real edges + N self loops) ----------------
__global__ void edge_max_kernel(const long long* __restrict__ ei, int E, int N, int H,
                                const float* __restrict__ a_src,
                                const float* __restrict__ a_dst,
                                float* __restrict__ e_buf,
                                float* __restrict__ emax) {
  int i = blockIdx.x * blockDim.x + threadIdx.x;
  if (i >= E + N) return;
  int s, d;
  if (i < E) { s = (int)ei[i]; d = (int)ei[(long long)E + i]; }
  else       { s = d = i - E; }
  for (int hh = 0; hh < H; ++hh) {
    float e = a_src[s * H + hh] + a_dst[d * H + hh];
    e = (e > 0.f) ? e : 0.2f * e;             // leaky_relu, slope 0.2
    e_buf[(long long)i * H + hh] = e;
    atomicMaxF32(&emax[d * H + hh], e);
  }
}

__global__ void edge_expsum_kernel(const long long* __restrict__ ei, int E, int N, int H,
                                   const float* __restrict__ emax,
                                   float* __restrict__ e_buf,
                                   float* __restrict__ denom) {
  int i = blockIdx.x * blockDim.x + threadIdx.x;
  if (i >= E + N) return;
  int d;
  if (i < E) d = (int)ei[(long long)E + i];
  else       d = i - E;
  for (int hh = 0; hh < H; ++hh) {
    float e = expf(e_buf[(long long)i * H + hh] - emax[d * H + hh]);
    e_buf[(long long)i * H + hh] = e;
    atomicAdd(&denom[d * H + hh], e);
  }
}

// One block per edge, F = H*C threads: out[d, f] += alpha[i, f/C] * h[s, f]
__global__ void edge_scatter_kernel(const long long* __restrict__ ei, int E, int N,
                                    int H, int C,
                                    const float* __restrict__ e_buf,
                                    const float* __restrict__ denom,
                                    const float* __restrict__ h,
                                    float* __restrict__ out) {
  int i = blockIdx.x;
  int f = threadIdx.x;
  int F = H * C;
  int s, d;
  if (i < E) { s = (int)ei[i]; d = (int)ei[(long long)E + i]; }
  else       { s = d = i - E; }
  int hh = f / C;
  float alpha = e_buf[(long long)i * H + hh] / denom[d * H + hh];
  atomicAdd(&out[(long long)d * F + f], alpha * h[(long long)s * F + f]);
}

// out1 = gelu(out1 + b1) (exact), plus bf16 copy for layer-2 GEMM
__global__ void bias_gelu_cvt_kernel(float* __restrict__ io,
                                     const float* __restrict__ bias,
                                     unsigned short* __restrict__ out_bf,
                                     long long total, int F) {
  long long i = (long long)blockIdx.x * blockDim.x + threadIdx.x;
  if (i >= total) return;
  float v = io[i] + bias[(int)(i % F)];
  float g = 0.5f * v * (1.0f + erff(v * 0.70710678118654752f));
  io[i] = g;
  out_bf[i] = f32_to_bf16(g);
}

__global__ void bias_add_kernel(float* __restrict__ io,
                                const float* __restrict__ bias,
                                long long total, int F) {
  long long i = (long long)blockIdx.x * blockDim.x + threadIdx.x;
  if (i >= total) return;
  io[i] += bias[(int)(i % F)];
}

// ---------------- launch ----------------
extern "C" void kernel_launch(void* const* d_in, const int* in_sizes, int n_in,
                              void* d_out, int out_size, void* d_ws, size_t ws_size,
                              hipStream_t stream) {
  const float*     x        = (const float*)d_in[0];
  const long long* ei       = (const long long*)d_in[1];   // int64 edge_index [2,E]
  const float*     W1       = (const float*)d_in[2];
  const float*     att_src1 = (const float*)d_in[3];
  const float*     att_dst1 = (const float*)d_in[4];
  const float*     b1       = (const float*)d_in[5];
  const float*     W2       = (const float*)d_in[6];
  const float*     att_src2 = (const float*)d_in[7];
  const float*     att_dst2 = (const float*)d_in[8];
  const float*     b2       = (const float*)d_in[9];
  float* out = (float*)d_out;

  const int Fin = 256;
  const int N   = in_sizes[0] / Fin;      // 20000
  const int E   = in_sizes[1] / 2;        // 320000
  const int F1  = in_sizes[5];            // 512 = H1*C1
  const int H1  = 4, C1 = F1 / 4;         // 4, 128
  const int F2  = in_sizes[9];            // 256
  const int H2  = 1, C2 = F2;             // 1, 256
  const int EN  = E + N;

  // ---- workspace layout ----
  char* ws = (char*)d_ws;
  size_t off = 0;
  auto alloc = [&](size_t bytes) {
    char* p = ws + off;
    off += (bytes + 255) & ~(size_t)255;
    return p;
  };
  float*          h1     = (float*)alloc((size_t)N * F1 * 4);
  float*          out1   = (float*)alloc((size_t)N * F1 * 4);
  unsigned short* xbf    = (unsigned short*)alloc((size_t)N * Fin * 2);
  unsigned short* h1bf   = (unsigned short*)alloc((size_t)N * F1 * 2);
  unsigned short* w1p    = (unsigned short*)alloc((size_t)Fin * F1 * 2);
  unsigned short* w2p    = (unsigned short*)alloc((size_t)F1 * F2 * 2);
  float*          h2     = (float*)alloc((size_t)N * F2 * 4);
  float*          a_src  = (float*)alloc((size_t)N * H1 * 4);
  float*          a_dst  = (float*)alloc((size_t)N * H1 * 4);
  float*          emax   = (float*)alloc((size_t)N * H1 * 4);
  float*          denom  = (float*)alloc((size_t)N * H1 * 4);
  float*          e_buf  = (float*)alloc((size_t)EN * H1 * 4);
  (void)ws_size; (void)n_in; (void)out_size;

  const int TB = 256;
  auto blocks = [&](long long n) { return (unsigned)((n + TB - 1) / TB); };
  const float NEG_INF = -__builtin_inff();

  // ---- prep ----
  cvt_bf16_kernel<<<blocks((long long)N * Fin), TB, 0, stream>>>(x, xbf, (long long)N * Fin);
  pack_w_kernel<<<blocks((long long)Fin * F1), TB, 0, stream>>>(W1, w1p, Fin, F1);
  pack_w_kernel<<<blocks((long long)F1 * F2), TB, 0, stream>>>(W2, w2p, F1, F2);

  // ---- layer 1 ----
  {
    dim3 grid(N / 32, F1 / 256);          // 625 x 2
    gemm_bf16_wmma_kernel<<<grid, TB, 0, stream>>>(xbf, w1p, h1, N, F1, Fin);
  }
  attn_coeff_kernel<<<blocks((long long)N * H1), TB, 0, stream>>>(
      h1, att_src1, att_dst1, a_src, a_dst, N, H1, C1);
  fill_f32_kernel<<<blocks((long long)N * H1), TB, 0, stream>>>(emax, NEG_INF, (long long)N * H1);
  fill_f32_kernel<<<blocks((long long)N * H1), TB, 0, stream>>>(denom, 0.f, (long long)N * H1);
  fill_f32_kernel<<<blocks((long long)N * F1), TB, 0, stream>>>(out1, 0.f, (long long)N * F1);
  edge_max_kernel<<<blocks(EN), TB, 0, stream>>>(ei, E, N, H1, a_src, a_dst, e_buf, emax);
  edge_expsum_kernel<<<blocks(EN), TB, 0, stream>>>(ei, E, N, H1, emax, e_buf, denom);
  edge_scatter_kernel<<<EN, F1, 0, stream>>>(ei, E, N, H1, C1, e_buf, denom, h1, out1);
  bias_gelu_cvt_kernel<<<blocks((long long)N * F1), TB, 0, stream>>>(
      out1, b1, h1bf, (long long)N * F1, F1);

  // ---- layer 2 ----
  {
    dim3 grid(N / 32, F2 / 256);          // 625 x 1
    gemm_bf16_wmma_kernel<<<grid, TB, 0, stream>>>(h1bf, w2p, h2, N, F2, F1);
  }
  attn_coeff_kernel<<<blocks((long long)N * H2), TB, 0, stream>>>(
      h2, att_src2, att_dst2, a_src, a_dst, N, H2, C2);
  fill_f32_kernel<<<blocks((long long)N * H2), TB, 0, stream>>>(emax, NEG_INF, (long long)N * H2);
  fill_f32_kernel<<<blocks((long long)N * H2), TB, 0, stream>>>(denom, 0.f, (long long)N * H2);
  fill_f32_kernel<<<blocks((long long)N * F2), TB, 0, stream>>>(out, 0.f, (long long)N * F2);
  edge_max_kernel<<<blocks(EN), TB, 0, stream>>>(ei, E, N, H2, a_src, a_dst, e_buf, emax);
  edge_expsum_kernel<<<blocks(EN), TB, 0, stream>>>(ei, E, N, H2, emax, e_buf, denom);
  edge_scatter_kernel<<<EN, F2, 0, stream>>>(ei, E, N, H2, C2, e_buf, denom, h2, out);
  bias_add_kernel<<<blocks((long long)N * F2), TB, 0, stream>>>(out, b2, (long long)N * F2, F2);
}
